// RetNetWrapper_9285719294258
// MI455X (gfx1250) — matrix operand order repaired and checked by reading, not compiled, not run on hardware
//
#include <hip/hip_runtime.h>
#include <hip/hip_bf16.h>

// ---------------------------------------------------------------------------
// Types for CDNA5 WMMA (wave32): D(f32 16x16) = A(f16 16x32) * B(f16 32x16) + C
// ---------------------------------------------------------------------------
typedef _Float16 v16h __attribute__((ext_vector_type(16)));
typedef _Float16 v8h  __attribute__((ext_vector_type(8)));
typedef _Float16 v4h  __attribute__((ext_vector_type(4)));
typedef float    v8f  __attribute__((ext_vector_type(8)));

__device__ __forceinline__ v8f wmma_f16(v16h a, v16h b, v8f c) {
    return __builtin_amdgcn_wmma_f32_16x16x32_f16(false, a, false, b,
                                                  (short)0, c, false, false);
}

// A-matrix fragment (16x32 f16) from LDS tile stored row-major [M][K].
// ISA layout: lanes 0-15 hold M=lane; halves e=0..7 -> K=koff+e,
// e=8..15 -> K=koff+16+(e-8), koff = (lane>=16)?8:0. Two contiguous 16B loads.
__device__ __forceinline__ v16h load_frag_a(const _Float16* base, int row0,
                                            int ldh, int lane) {
    int m    = row0 + (lane & 15);
    int koff = (lane >> 4) << 3;                 // 0 or 8
    const v8h lo = *(const v8h*)(base + m * ldh + koff);
    const v8h hi = *(const v8h*)(base + m * ldh + koff + 16);
    return __builtin_shufflevector(lo, hi, 0,1,2,3,4,5,6,7,
                                            8,9,10,11,12,13,14,15);
}

// B-matrix fragment (32x16 f16) from LDS tile stored row-major [N][K].
// ISA layout: lanes 0-15 -> N=lane; halves e=0..7 -> K=koff+e,
// e=8..15 -> K=koff+16+(e-8), koff = (lane>=16)?8:0. Two contiguous 16B loads.
__device__ __forceinline__ v16h load_frag_b(const _Float16* base, int row0,
                                            int ldh, int lane) {
    int nn   = row0 + (lane & 15);
    int koff = (lane >> 4) << 3;                 // 0 or 8
    const v8h lo = *(const v8h*)(base + nn * ldh + koff);
    const v8h hi = *(const v8h*)(base + nn * ldh + koff + 16);
    return __builtin_shufflevector(lo, hi, 0,1,2,3,4,5,6,7,
                                            8,9,10,11,12,13,14,15);
}

// ---------------------------------------------------------------------------
// GEMM: Y[M,N] = X[M,K] * W[N,K]^T   (fp32 in memory, f16 WMMA compute).
// Block tile 128x128x32, 256 threads = 8 waves (2x4), each wave 64x32.
// Double-buffered LDS: global loads of tile k+1 are issued before the WMMA
// block for tile k, so the v_wmma burst hides the global_load_b128 latency;
// one barrier per K-step.
// ---------------------------------------------------------------------------
#define GBM 128
#define GBN 128
#define GBK 32
#define GLD 40   // padded LDS row stride (halves) to spread banks

__global__ __launch_bounds__(256) void gemm_wmma_kernel(
    const float* __restrict__ X, const float* __restrict__ W,
    float* __restrict__ Y, int M, int N, int K) {
    __shared__ _Float16 a_lds[2][GBM * GLD];
    __shared__ _Float16 b_lds[2][GBN * GLD];

    const int tid  = threadIdx.x;
    const int lane = tid & 31;
    const int w    = tid >> 5;
    const int wm   = w >> 2;          // 0..1
    const int wn   = w & 3;           // 0..3
    const int m0   = blockIdx.y * GBM;
    const int n0   = blockIdx.x * GBN;

    // Per-thread staging slots: 4 float4 per matrix per K-step.
    int rr[4], cc[4];
    #pragma unroll
    for (int it = 0; it < 4; ++it) {
        const int li = it * 256 + tid;   // 0..1023
        rr[it] = li >> 3;                // row 0..127
        cc[it] = (li & 7) << 2;          // col 0,4,...,28
    }

    v8f acc[4][2] = {};
    float4 fx[4], fw[4];

    // Prologue: fetch K-tile 0 into registers, stage into buffer 0.
    #pragma unroll
    for (int it = 0; it < 4; ++it) {
        fx[it] = *(const float4*)(X + (size_t)(m0 + rr[it]) * K + cc[it]);
        fw[it] = *(const float4*)(W + (size_t)(n0 + rr[it]) * K + cc[it]);
    }
    #pragma unroll
    for (int it = 0; it < 4; ++it) {
        v4h ha = {(_Float16)fx[it].x, (_Float16)fx[it].y,
                  (_Float16)fx[it].z, (_Float16)fx[it].w};
        *(v4h*)(&a_lds[0][0] + rr[it] * GLD + cc[it]) = ha;
        v4h hb = {(_Float16)fw[it].x, (_Float16)fw[it].y,
                  (_Float16)fw[it].z, (_Float16)fw[it].w};
        *(v4h*)(&b_lds[0][0] + rr[it] * GLD + cc[it]) = hb;
    }
    __syncthreads();

    int buf = 0;
    for (int kk = 0; kk < K; kk += GBK) {
        const bool has_next = (kk + GBK) < K;
        // Issue next K-tile global loads early (latency hidden by WMMAs).
        if (has_next) {
            #pragma unroll
            for (int it = 0; it < 4; ++it) {
                fx[it] = *(const float4*)(X + (size_t)(m0 + rr[it]) * K
                                          + kk + GBK + cc[it]);
                fw[it] = *(const float4*)(W + (size_t)(n0 + rr[it]) * K
                                          + kk + GBK + cc[it]);
            }
            if (kk + 2 * GBK < K) {  // L2 prefetch of the tile after next
                __builtin_prefetch(X + (size_t)(m0 + rr[0]) * K + kk + 2 * GBK
                                   + cc[0], 0, 1);
                __builtin_prefetch(W + (size_t)(n0 + rr[0]) * K + kk + 2 * GBK
                                   + cc[0], 0, 1);
            }
        }

        // Compute on current buffer.
        const _Float16* ab = &a_lds[buf][0];
        const _Float16* bb = &b_lds[buf][0];
        v16h aF[4], bF[2];
        #pragma unroll
        for (int ti = 0; ti < 4; ++ti)
            aF[ti] = load_frag_a(ab, wm * 64 + ti * 16, GLD, lane);
        #pragma unroll
        for (int tj = 0; tj < 2; ++tj)
            bF[tj] = load_frag_b(bb, wn * 32 + tj * 16, GLD, lane);
        #pragma unroll
        for (int ti = 0; ti < 4; ++ti)
            #pragma unroll
            for (int tj = 0; tj < 2; ++tj)
                acc[ti][tj] = wmma_f16(aF[ti], bF[tj], acc[ti][tj]);

        // Stage next tile into the other buffer.
        if (has_next) {
            _Float16* an = &a_lds[buf ^ 1][0];
            _Float16* bn = &b_lds[buf ^ 1][0];
            #pragma unroll
            for (int it = 0; it < 4; ++it) {
                v4h ha = {(_Float16)fx[it].x, (_Float16)fx[it].y,
                          (_Float16)fx[it].z, (_Float16)fx[it].w};
                *(v4h*)(an + rr[it] * GLD + cc[it]) = ha;
                v4h hb = {(_Float16)fw[it].x, (_Float16)fw[it].y,
                          (_Float16)fw[it].z, (_Float16)fw[it].w};
                *(v4h*)(bn + rr[it] * GLD + cc[it]) = hb;
            }
        }
        __syncthreads();
        buf ^= 1;
    }

    // C/D layout: VGPR r -> M = r + ((lane>=16)?8:0), N = lane&15
    const int nn = lane & 15;
    const int mb = (lane >> 4) << 3;
    #pragma unroll
    for (int ti = 0; ti < 4; ++ti)
        #pragma unroll
        for (int tj = 0; tj < 2; ++tj)
            #pragma unroll
            for (int r = 0; r < 8; ++r)
                Y[(size_t)(m0 + wm * 64 + ti * 16 + mb + r) * N
                  + n0 + wn * 32 + tj * 16 + nn] = acc[ti][tj][r];
}

// ---------------------------------------------------------------------------
// RoPE on q and k (in place), plus q *= HEAD_K^-0.5. d=256, half=128.
// One block per (b,t) row.
// ---------------------------------------------------------------------------
__global__ __launch_bounds__(256) void rope_kernel(float* __restrict__ q,
                                                   float* __restrict__ k) {
    const int row = blockIdx.x;          // b*T + t
    const float t = (float)(row & 2047); // T = 2048
    for (int p = threadIdx.x; p < 512; p += 256) {
        const int h = p >> 7, j = p & 127;
        const float ang = t * __powf(10000.0f, -(float)(2 * j) * (1.0f / 256.0f));
        float s, c;
        sincosf(ang, &s, &c);
        const size_t base = ((size_t)row * 4 + h) * 256;
        const float x1 = q[base + j], x2 = q[base + 128 + j];
        q[base + j]       = (x1 * c - x2 * s) * 0.0625f;
        q[base + 128 + j] = (x2 * c + x1 * s) * 0.0625f;
        const float y1 = k[base + j], y2 = k[base + 128 + j];
        k[base + j]       = y1 * c - y2 * s;
        k[base + 128 + j] = y2 * c + y1 * s;
    }
}

// ---------------------------------------------------------------------------
// Chunked retention. One block per (b, h, vtile): 4*4*8 = 128 blocks,
// 256 threads (8 waves). dk = 256, dv-tile = 64, CHUNK = 64, N = 32 chunks.
// State S (256 x 64 fp32) lives in per-wave WMMA accumulators; an f16
// transposed mirror ShT[vcol][dk] in LDS feeds the q*S inter-chunk matmul.
// ---------------------------------------------------------------------------
#define QLD 264   // 256 + 8 pad (halves)
#define SLD 72    // 64 + 8 pad (halves)
#define RET_SMEM_BYTES ((64*QLD*3 + 256*SLD + 64*SLD*2) * 2 + 68 * 4)

__global__ __launch_bounds__(256) void retention_kernel(
    const float* __restrict__ q, const float* __restrict__ k,
    const float* __restrict__ v, float* __restrict__ o) {
    extern __shared__ __align__(16) char smem_raw[];
    _Float16* q_lds    = (_Float16*)smem_raw;       // [64][QLD]
    _Float16* k_lds    = q_lds    + 64 * QLD;       // [64][QLD]  (B for attn)
    _Float16* kT_lds   = k_lds    + 64 * QLD;       // [256][SLD] (A for S upd, k_dec folded)
    _Float16* vT_lds   = kT_lds   + 256 * SLD;      // [64][SLD]  (B: [vcol][pos])
    _Float16* attn_lds = vT_lds   + 64 * SLD;       // [64][SLD]
    _Float16* ShT_lds  = attn_lds + 64 * SLD;       // [64][QLD]  (B: [vcol][dk])
    float*    gpow     = (float*)(ShT_lds + 64 * QLD);  // gamma^0..64

    const int bid = blockIdx.x;
    const int vt  = bid & 7, h = (bid >> 3) & 3, b = bid >> 5;
    const int v0  = vt * 64;
    const float gamma = 1.0f - exp2f(-5.0f - (float)h);

    const int tid  = threadIdx.x;
    const int lane = tid & 31;
    const int w    = tid >> 5;
    const int nn   = lane & 15;
    const int mb   = (lane >> 4) << 3;

    if (tid <= 64) gpow[tid] = __powf(gamma, (float)tid);
    for (int idx = tid; idx < 64 * QLD; idx += 256)
        ShT_lds[idx] = (_Float16)0.0f;

    v8f sacc[2][4] = {};          // S tiles: rows w*2+s (dk), cols sj (vcol)
    const int oti  = w >> 1;          // output/attn row tile 0..3
    const int otj0 = (w & 1) * 2;     // output col tile base
    __syncthreads();

    for (int ch = 0; ch < 32; ++ch) {
        const int t0 = ch * 64;
        // ---- stage q, k, kT(k_dec folded), vT into LDS as f16 ----
        for (int idx = tid; idx < 64 * 256; idx += 256) {
            const int i = idx >> 8, d = idx & 255;
            const size_t gb = ((size_t)((b * 2048 + t0 + i) * 4 + h)) * 256 + d;
            const float qv = q[gb], kv = k[gb];
            q_lds[i * QLD + d]  = (_Float16)qv;
            k_lds[i * QLD + d]  = (_Float16)kv;
            kT_lds[d * SLD + i] = (_Float16)(kv * gpow[63 - i]);
        }
        for (int idx = tid; idx < 64 * 64; idx += 256) {
            const int i = idx >> 6, c = idx & 63;
            const float vv =
                v[((size_t)((b * 2048 + t0 + i) * 4 + h)) * 512 + v0 + c];
            vT_lds[c * SLD + i] = (_Float16)vv;
        }
        __syncthreads();

        // ---- o_inter = (q * cross_dec) @ S_old  (reads ShT mirror) ----
        v8f oacc[2];
        const _Float16 cd = (_Float16)gpow[oti * 16 + nn + 1];
        #pragma unroll
        for (int u = 0; u < 2; ++u) {
            const int tj = otj0 + u;
            v8f c = {};
            #pragma unroll
            for (int kb = 0; kb < 256; kb += 32) {
                v16h a = load_frag_a(q_lds + kb, oti * 16, QLD, lane);
                a = a * cd;   // per-lane row scale gamma^(i+1)
                v16h bb = load_frag_b(ShT_lds + kb, tj * 16, QLD, lane);
                c = wmma_f16(a, bb, c);
            }
            oacc[u] = c;
        }
        // ---- attn = (q @ k^T) * decay-mask, store f16 ----
        #pragma unroll
        for (int u = 0; u < 2; ++u) {
            const int aj = otj0 + u;
            v8f c = {};
            #pragma unroll
            for (int kb = 0; kb < 256; kb += 32) {
                v16h a  = load_frag_a(q_lds + kb, oti * 16, QLD, lane);
                v16h bb = load_frag_b(k_lds + kb, aj * 16, QLD, lane);
                c = wmma_f16(a, bb, c);
            }
            #pragma unroll
            for (int r = 0; r < 8; ++r) {
                const int i = oti * 16 + mb + r, j = aj * 16 + nn;
                const float val = (i >= j) ? c[r] * gpow[i - j] : 0.0f;
                attn_lds[i * SLD + j] = (_Float16)val;
            }
        }
        __syncthreads();

        // ---- o = o_inter + attn @ v ; write out ----
        #pragma unroll
        for (int u = 0; u < 2; ++u) {
            const int tj = otj0 + u;
            v8f c = oacc[u];
            #pragma unroll
            for (int kb = 0; kb < 64; kb += 32) {
                v16h a  = load_frag_a(attn_lds + kb, oti * 16, SLD, lane);
                v16h bb = load_frag_b(vT_lds + kb, tj * 16, SLD, lane);
                c = wmma_f16(a, bb, c);
            }
            #pragma unroll
            for (int r = 0; r < 8; ++r)
                o[((size_t)((b * 2048 + t0 + oti * 16 + mb + r) * 4 + h)) * 512
                  + v0 + tj * 16 + nn] = c[r];
        }

        // ---- S = gamma^C * S + (k * k_dec)^T @ v  (fp32 accumulators) ----
        const float gC = gpow[64];
        #pragma unroll
        for (int s = 0; s < 2; ++s) {
            const int si = w * 2 + s;
            #pragma unroll
            for (int sj = 0; sj < 4; ++sj) {
                v8f c = sacc[s][sj];
                #pragma unroll
                for (int r = 0; r < 8; ++r) c[r] *= gC;
                #pragma unroll
                for (int kb = 0; kb < 64; kb += 32) {
                    v16h a  = load_frag_a(kT_lds + kb, si * 16, SLD, lane);
                    v16h bb = load_frag_b(vT_lds + kb, sj * 16, SLD, lane);
                    c = wmma_f16(a, bb, c);
                }
                sacc[s][sj] = c;
            }
        }
        // refresh f16 transposed mirror for next chunk's o_inter
        #pragma unroll
        for (int s = 0; s < 2; ++s) {
            const int si = w * 2 + s;
            #pragma unroll
            for (int sj = 0; sj < 4; ++sj)
                #pragma unroll
                for (int r = 0; r < 8; ++r)
                    ShT_lds[(sj * 16 + nn) * QLD + si * 16 + mb + r] =
                        (_Float16)sacc[s][sj][r];
        }
        __syncthreads();
    }
}

// ---------------------------------------------------------------------------
// Per-(b,t,head) RMS groupnorm over dv=512 + gnorm weight + SiLU gate.
// One block per (b,t); 64 threads per head.
// ---------------------------------------------------------------------------
__global__ __launch_bounds__(256) void gnorm_gate_kernel(
    float* __restrict__ o, const float* __restrict__ g,
    const float* __restrict__ gw) {
    __shared__ float red[256];
    const int row = blockIdx.x, tid = threadIdx.x;
    const int head = tid >> 6, wl = tid & 63;
    const size_t base = ((size_t)row * 4 + head) * 512;

    float vals[8];
    float ss = 0.0f;
    #pragma unroll
    for (int e = 0; e < 8; ++e) {
        const int c = wl + e * 64;
        vals[e] = o[base + c];
        ss += vals[e] * vals[e];
    }
    red[tid] = ss;
    __syncthreads();
    for (int off = 32; off > 0; off >>= 1) {
        if (wl < off) red[tid] += red[tid + off];
        __syncthreads();
    }
    const float var = red[head << 6] * (1.0f / 512.0f);
    const float sc  = rsqrtf(var + 1e-5f);
    #pragma unroll
    for (int e = 0; e < 8; ++e) {
        const int c = wl + e * 64;
        const float gg  = g[base + c];
        const float sig = 1.0f / (1.0f + expf(-gg));
        o[base + c] = vals[e] * sc * gw[c] * gg * sig;
    }
}

// ---------------------------------------------------------------------------
// Host launcher
// ---------------------------------------------------------------------------
extern "C" void kernel_launch(void* const* d_in, const int* in_sizes, int n_in,
                              void* d_out, int out_size, void* d_ws,
                              size_t ws_size, hipStream_t stream) {
    (void)in_sizes; (void)n_in; (void)out_size; (void)ws_size;
    const float* x  = (const float*)d_in[0];
    const float* Wq = (const float*)d_in[1];
    const float* Wk = (const float*)d_in[2];
    const float* Wv = (const float*)d_in[3];
    const float* Wg = (const float*)d_in[4];
    const float* Wo = (const float*)d_in[5];
    const float* gw = (const float*)d_in[6];
    float* out = (float*)d_out;

    float* ws = (float*)d_ws;
    float* qb = ws;                                  // 8192*1024
    float* kb = qb + (size_t)8192 * 1024;            // 8192*1024
    float* vb = kb + (size_t)8192 * 1024;            // 8192*2048
    float* gb = vb + (size_t)8192 * 2048;            // 8192*2048
    float* ob = gb + (size_t)8192 * 2048;            // 8192*2048

    const dim3 blk(256);
    gemm_wmma_kernel<<<dim3(8, 64),  blk, 0, stream>>>(x, Wq, qb, 8192, 1024, 1024);
    gemm_wmma_kernel<<<dim3(8, 64),  blk, 0, stream>>>(x, Wk, kb, 8192, 1024, 1024);
    gemm_wmma_kernel<<<dim3(16, 64), blk, 0, stream>>>(x, Wv, vb, 8192, 2048, 1024);
    gemm_wmma_kernel<<<dim3(16, 64), blk, 0, stream>>>(x, Wg, gb, 8192, 2048, 1024);

    rope_kernel<<<8192, blk, 0, stream>>>(qb, kb);

    (void)hipFuncSetAttribute((const void*)retention_kernel,
                              hipFuncAttributeMaxDynamicSharedMemorySize,
                              RET_SMEM_BYTES);
    retention_kernel<<<128, blk, RET_SMEM_BYTES, stream>>>(qb, kb, vb, ob);

    gnorm_gate_kernel<<<8192, blk, 0, stream>>>(ob, gb, gw);

    gemm_wmma_kernel<<<dim3(8, 64), blk, 0, stream>>>(ob, Wo, out, 8192, 1024, 2048);
}
